// PDFNetTruthChildren_57982058496298
// MI455X (gfx1250) — matrix operand order repaired and checked by reading, not compiled, not run on hardware
//
#include <hip/hip_runtime.h>
#include <hip/hip_bf16.h>

// ---------------------------------------------------------------------------
// PDFNetTruthChildren for MI455X (gfx1250, wave32, WMMA).
//
//   k_zero              : zero node_bias accumulator in workspace
//   k_w2_transpose_bf16 : mass.l2.w f32[1024k][1024n] -> bf16 W2T[n][k]
//   k_collapse_affine x4: fold each (no-activation) enc+dec chain to O=a*x+c
//   k_node_outputs      : O_eta/O_energy/O_phi/O_pT elementwise (memory bound)
//   k_edge_mlp_wmma     : relu MLP on E_T via v_wmma_f32_16x16x32_bf16,
//                         4-N-tile groups amortize A-fragment LDS loads,
//                         fused relu + w3 dot + atomic segment-sum
//   k_edge_index_out    : O_Index[e] = bias[dst[e]] - bias[src[e]]
// ---------------------------------------------------------------------------

#define N_NODES 50000
#define N_EDGES 200000

typedef __bf16 bf16_t;
typedef __attribute__((ext_vector_type(16))) __bf16 v16bf;
typedef __attribute__((ext_vector_type(8)))  __bf16 bfx8;
typedef __attribute__((ext_vector_type(2)))  __bf16 bfx2;
typedef __attribute__((ext_vector_type(8)))  float  v8f;

union ABFrag { bfx8 h[2]; v16bf v; };

// ------------------------------ tiny kernels -------------------------------

__global__ __launch_bounds__(256) void k_zero(float* p, int n) {
  int i = blockIdx.x * blockDim.x + threadIdx.x;
  if (i < n) p[i] = 0.f;
}

// W2 is row-major [k=1024][n=1024] f32; produce W2T row-major [n][k] bf16 so
// each WMMA B-fragment lane reads two runs of 8 contiguous bf16 of one output column.
__global__ __launch_bounds__(256) void k_w2_transpose_bf16(
    const float* __restrict__ w2, bf16_t* __restrict__ w2t) {
  int idx = blockIdx.x * blockDim.x + threadIdx.x;   // 1M threads
  int k = idx >> 10;
  int n = idx & 1023;
  w2t[n * 1024 + k] = (bf16_t)w2[k * 1024 + n];      // coalesced read
}

// ------------------- affine collapse of the linear chains ------------------
// chain: x[N,1] -> l1(1,256) -> l2(256,1024) -> l3(1024,256) -> l4(256,1)
// No activations => O = a*x + c with scalar a,c derived from the weights.

struct ChainPtrs {
  const float* ew1;  // enc.l1.w [1x256]
  const float* eb1;  // enc.l1.b [256]
  const float* eW2;  // enc.l2.w [256x1024]
  const float* eb2;  // enc.l2.b [1024]
  const float* dW1;  // dec.l1.w [1024x256]
  const float* db1;  // dec.l1.b [256]
  const float* dw2;  // dec.l2.w [256x1]
  const float* db2;  // dec.l2.b [1]
};

__global__ __launch_bounds__(256) void k_collapse_affine(ChainPtrs cp,
                                                         float* coeff_out) {
  __shared__ float sA[1024];
  __shared__ float sB[1024];
  __shared__ float red[2][256];
  const int tid = threadIdx.x;

  // stage 1: aV[o] = w1 . W2[:,o] ; bV[o] = b1 . W2[:,o] + b2[o]
  for (int oo = 0; oo < 4; ++oo) {
    int o = tid + oo * 256;
    float accA = 0.f, accB = 0.f;
    for (int j = 0; j < 256; ++j) {
      float wv = cp.eW2[j * 1024 + o];   // consecutive o -> coalesced
      accA = fmaf(cp.ew1[j], wv, accA);
      accB = fmaf(cp.eb1[j], wv, accB);
    }
    sA[o] = accA;
    sB[o] = accB + cp.eb2[o];
  }
  __syncthreads();

  // stage 2: push through dec.l1 (1024->256), then dot with dec.l2 (256->1)
  float accA = 0.f, accB = 0.f;
  for (int o = 0; o < 1024; ++o) {
    float wv = cp.dW1[o * 256 + tid];    // consecutive tid -> coalesced
    accA = fmaf(sA[o], wv, accA);
    accB = fmaf(sB[o], wv, accB);
  }
  accB += cp.db1[tid];
  float w4 = cp.dw2[tid];
  red[0][tid] = accA * w4;
  red[1][tid] = accB * w4;
  __syncthreads();
  for (int s = 128; s > 0; s >>= 1) {
    if (tid < s) {
      red[0][tid] += red[0][tid + s];
      red[1][tid] += red[1][tid + s];
    }
    __syncthreads();
  }
  if (tid == 0) {
    coeff_out[0] = red[0][0];               // a
    coeff_out[1] = red[1][0] + cp.db2[0];   // c
  }
}

// coeffs = [a_eta,c_eta, a_en,c_en, a_phi,c_phi, a_pt,c_pt]
__global__ __launch_bounds__(256) void k_node_outputs(
    const float* __restrict__ n_eta, const float* __restrict__ n_en,
    const float* __restrict__ n_pt, const float* __restrict__ n_phi,
    const float* __restrict__ coeffs, float* __restrict__ out) {
  int i = blockIdx.x * blockDim.x + threadIdx.x;
  if (i >= N_NODES) return;
  out[0 * N_NODES + i] = fmaf(coeffs[0], n_eta[i], coeffs[1]);  // O_eta
  out[1 * N_NODES + i] = fmaf(coeffs[2], n_en[i],  coeffs[3]);  // O_energy
  out[2 * N_NODES + i] = fmaf(coeffs[4], n_phi[i], coeffs[5]);  // O_phi
  out[3 * N_NODES + i] = fmaf(coeffs[6], n_pt[i],  coeffs[7]);  // O_pT
}

// ----------------------- mass tagger: WMMA edge MLP ------------------------
// sel[e] = b3 + sum_n w3[n] * relu( b2[n] + sum_k W2[k,n]*relu(E_T[e]*w1[k]+b1[k]) )
// 8 waves/block, each wave owns a 16-edge M-tile; block covers 128 edges.
// N-tiles processed in groups of 4 so each A-fragment LDS load feeds 4 WMMAs.

#define MTILE 128
#define HPAD  1032                       // 1024 + 8 bf16: shifts 4 banks/row
#define EDGE_SMEM_BYTES (MTILE * HPAD * 2)

__global__ __launch_bounds__(256) void k_edge_mlp_wmma(
    const float* __restrict__ et,        // [E]
    const int*   __restrict__ eidx,      // [2*E], row0 = src
    const float* __restrict__ w1,        // [1024] mass.l1.w
    const float* __restrict__ b1,        // [1024] mass.l1.b
    const bf16_t* __restrict__ w2t,      // [1024n][1024k] bf16
    const float* __restrict__ b2,        // [1024] mass.l2.b
    const float* __restrict__ w3,        // [1024] mass.l3.w
    const float* __restrict__ b3,        // [1]    mass.l3.b
    float* __restrict__ node_bias) {     // [N] accumulator (pre-zeroed)
  extern __shared__ __align__(16) char smem_raw[];
  bf16_t* h1 = (bf16_t*)smem_raw;        // [MTILE][HPAD] bf16

  const int tid    = threadIdx.x;
  const int wave   = tid >> 5;
  const int lane   = tid & 31;
  const int eBlock = blockIdx.x * MTILE;

  // ---- stage h1 = relu(E_T*w1 + b1) -> LDS as bf16, two K per thread ----
  for (int idx = tid; idx < MTILE * 512; idx += 256) {
    int row = idx >> 9;
    int k2  = (idx & 511) * 2;
    int e   = eBlock + row;
    float x = (e < N_EDGES) ? et[e] : 0.f;   // tail rows masked at write-out
    float2 wv = *(const float2*)(w1 + k2);   // global_load_b64
    float2 bv = *(const float2*)(b1 + k2);
    bfx2 pk;
    pk.x = (bf16_t)fmaxf(fmaf(x, wv.x, bv.x), 0.f);
    pk.y = (bf16_t)fmaxf(fmaf(x, wv.y, bv.y), 0.f);
    *(bfx2*)(h1 + row * HPAD + k2) = pk;     // ds_store_b32
  }
  __syncthreads();

  // ---- per-wave 16x1024 @ 1024x1024 with v_wmma_f32_16x16x32_bf16 ----
  const int  rowBase = wave * 16;
  const int  m16     = lane & 15;
  const bool hiHalf  = lane >= 16;
  // A lane layout (ISA 7.12.2, 16-bit A 16x32): lane<16 holds K {0..7,16..23},
  // lane>=16 holds K {8..15,24..31} of its row.
  const bf16_t* aRow = h1 + (rowBase + m16) * HPAD + (hiHalf ? 8 : 0);

  float pacc[8];
#pragma unroll
  for (int i = 0; i < 8; ++i) pacc[i] = 0.f;

  for (int ntg = 0; ntg < 16; ++ntg) {     // 4 N-tiles (64 columns) per group
    const int ncol0 = ntg * 64 + m16;
    // B lane layout (32x16 16-bit B): lane%16 = N, lane<16 holds K {0..7,16..23}, lanes>=16 hold K {8..15,24..31}.
    const bf16_t* bBase = w2t + ncol0 * 1024 + (hiHalf ? 8 : 0);
    if (ntg + 1 < 16) __builtin_prefetch(bBase + 64 * 1024, 0, 1);

    v8f cacc[4];
#pragma unroll
    for (int t = 0; t < 4; ++t)
      cacc[t] = (v8f){0.f, 0.f, 0.f, 0.f, 0.f, 0.f, 0.f, 0.f};

#pragma unroll 2
    for (int kk = 0; kk < 1024; kk += 32) {
      ABFrag A;
      A.h[0] = *(const bfx8*)(aRow + kk);        // ds_load_b128 (shared by 4)
      A.h[1] = *(const bfx8*)(aRow + kk + 16);   // ds_load_b128
#pragma unroll
      for (int t = 0; t < 4; ++t) {
        ABFrag B;
        B.h[0] = *(const bfx8*)(bBase + t * 16 * 1024 + kk);      // b128
        B.h[1] = *(const bfx8*)(bBase + t * 16 * 1024 + kk + 16);  // b128
        cacc[t] = __builtin_amdgcn_wmma_f32_16x16x32_bf16(
            /*neg_a=*/false, A.v, /*neg_b=*/false, B.v,
            /*c_mod=*/(short)0, cacc[t], /*reuse_a=*/false, /*reuse_b=*/false);
      }
    }

    // epilogue: relu(+b2) then dot with this column's w3 weight
#pragma unroll
    for (int t = 0; t < 4; ++t) {
      int   ncol = ncol0 + t * 16;
      float b2v  = b2[ncol];
      float w3v  = w3[ncol];
#pragma unroll
      for (int i = 0; i < 8; ++i) {
        float v = fmaxf(cacc[t][i] + b2v, 0.f);  // C: lane=N, vgpr i = M(+8)
        pacc[i] = fmaf(v, w3v, pacc[i]);
      }
    }
  }

  // reduce over the 16 lanes (16 N-columns) sharing each edge row
#pragma unroll
  for (int i = 0; i < 8; ++i) {
    float v = pacc[i];
    v += __shfl_xor(v, 1, 32);
    v += __shfl_xor(v, 2, 32);
    v += __shfl_xor(v, 4, 32);
    v += __shfl_xor(v, 8, 32);
    pacc[i] = v;
  }

  const float b3v = b3[0];
  if (lane == 0 || lane == 16) {          // lane0 -> M 0..7, lane16 -> M 8..15
    int mOff = hiHalf ? 8 : 0;
#pragma unroll
    for (int i = 0; i < 8; ++i) {
      int e = eBlock + rowBase + mOff + i;
      if (e < N_EDGES) {
        int src = eidx[e];                // flow target_to_source: index row 0
        atomicAdd(node_bias + src, pacc[i] + b3v);  // global_atomic_add_f32
      }
    }
  }
}

__global__ __launch_bounds__(256) void k_edge_index_out(
    const int* __restrict__ eidx, const float* __restrict__ node_bias,
    float* __restrict__ out_idx) {
  int e = blockIdx.x * blockDim.x + threadIdx.x;
  if (e >= N_EDGES) return;
  int src = eidx[e];
  int dst = eidx[N_EDGES + e];
  out_idx[e] = node_bias[dst] - node_bias[src];
}

// ------------------------------- launcher ----------------------------------

extern "C" void kernel_launch(void* const* d_in, const int* in_sizes, int n_in,
                              void* d_out, int out_size, void* d_ws,
                              size_t ws_size, hipStream_t stream) {
  (void)in_sizes; (void)n_in; (void)out_size; (void)ws_size;

  const float* N_eta    = (const float*)d_in[0];
  const float* N_energy = (const float*)d_in[1];
  const float* N_pT     = (const float*)d_in[2];
  const float* N_phi    = (const float*)d_in[3];
  const int*   eidx     = (const int*)d_in[4];
  const float* E_T      = (const float*)d_in[5];

  // params flattened jax-pytree style (sorted keys):
  //  6.. 9 delta_enc{l1.b,l1.w,l2.b,l2.w}   (dead code, unused)
  // 10..13 en_dec | 14..17 en_enc | 18..21 eta_dec | 22..25 eta_enc
  // 26..31 mass{l1.b,l1.w,l2.b,l2.w,l3.b,l3.w}
  // 32..35 phi_dec | 36..39 phi_enc | 40..43 pt_dec | 44..47 pt_enc
  auto F = [&](int i) { return (const float*)d_in[i]; };

  // workspace layout
  char*   ws        = (char*)d_ws;
  float*  node_bias = (float*)ws;                              // 50048 f32
  bf16_t* w2t       = (bf16_t*)(ws + 50048 * 4);               // 1M bf16 (2MB)
  float*  coeffs    = (float*)(ws + 50048 * 4 + 1024 * 1024 * 2); // 8 f32

  float* out       = (float*)d_out;
  float* out_index = out + 4 * N_NODES;

  // 1) zero segment-sum accumulator (every call: ws is not re-initialized)
  k_zero<<<(N_NODES + 255) / 256, 256, 0, stream>>>(node_bias, N_NODES);

  // 2) W2 -> bf16 transposed copy for WMMA B-fragments
  k_w2_transpose_bf16<<<(1024 * 1024) / 256, 256, 0, stream>>>(F(29), w2t);

  // 3) collapse the four activation-free chains to (a, c)
  ChainPtrs eta  = {F(23), F(22), F(25), F(24), F(19), F(18), F(21), F(20)};
  ChainPtrs ener = {F(15), F(14), F(17), F(16), F(11), F(10), F(13), F(12)};
  ChainPtrs phi  = {F(37), F(36), F(39), F(38), F(33), F(32), F(35), F(34)};
  ChainPtrs pt   = {F(45), F(44), F(47), F(46), F(41), F(40), F(43), F(42)};
  k_collapse_affine<<<1, 256, 0, stream>>>(eta,  coeffs + 0);
  k_collapse_affine<<<1, 256, 0, stream>>>(ener, coeffs + 2);
  k_collapse_affine<<<1, 256, 0, stream>>>(phi,  coeffs + 4);
  k_collapse_affine<<<1, 256, 0, stream>>>(pt,   coeffs + 6);

  // 4) node outputs (pure elementwise, HBM-bound, ~1.6MB total traffic)
  k_node_outputs<<<(N_NODES + 255) / 256, 256, 0, stream>>>(
      N_eta, N_energy, N_pT, N_phi, coeffs, out);

  // 5) mass tagger edge MLP: 0.42 TFLOP via bf16 WMMA, 258KB LDS A-staging
  (void)hipFuncSetAttribute((const void*)k_edge_mlp_wmma,
                            hipFuncAttributeMaxDynamicSharedMemorySize,
                            EDGE_SMEM_BYTES);
  int edgeBlocks = (N_EDGES + MTILE - 1) / MTILE;  // 1563
  k_edge_mlp_wmma<<<edgeBlocks, 256, EDGE_SMEM_BYTES, stream>>>(
      E_T, eidx, F(27), F(26), w2t, F(28), F(31), F(30), node_bias);

  // 6) O_Index = node_bias[dst] - node_bias[src]
  k_edge_index_out<<<(N_EDGES + 255) / 256, 256, 0, stream>>>(
      eidx, node_bias, out_index);
}
